// NTXentLoss_43482248904820
// MI455X (gfx1250) — matrix operand order repaired and checked by reading, not compiled, not run on hardware
//
#include <hip/hip_runtime.h>
#include <stdint.h>

// ---------------------------------------------------------------------------
// NT-Xent loss, N=8192, D=128, f32 input -> scalar f32.
// Compute-bound (17.2 GFLOP GEMM; 4MB input lives in L2). Strategy:
//   z = x/||x|| split into bf16 hi+lo; sim tile = 3x v_wmma_f32_16x16x32_bf16
//   (hi*hi + hi*lo + lo*hi, f32 accum) ~= f32 precision at bf16-WMMA rate.
//   Fixed-shift logsumexp (logits bounded by 1/T=10 since rows are unit-norm):
//   lse = 10 + ln(sum exp(val-10)) -> no running max, branchless hot loop,
//   never materialize the 8192x8192 sim matrix. Slab-0 B fragments are
//   software-pipelined across tiles to hide the per-iteration load latency.
// ---------------------------------------------------------------------------

typedef __attribute__((ext_vector_type(16))) __bf16 v16bf;
typedef __attribute__((ext_vector_type(8)))  float  v8f;

#define N_ROWS   8192
#define DIM      128
#define HALF_N   4096
#define N_TILES  (N_ROWS / 16)            // 512
// exp(val - 10) with val = c*10, computed as 2^(c*K1 - K1), K1 = 10*log2(e)
#define K1       14.4269504088896340736f

__device__ __forceinline__ uint16_t f32_to_bf16(float f) {
    union { float f; uint32_t u; } v; v.f = f;
    uint32_t u = v.u;
    u += 0x7FFFu + ((u >> 16) & 1u);     // round to nearest even
    return (uint16_t)(u >> 16);
}
__device__ __forceinline__ float bf16_to_f32(uint16_t h) {
    union { uint32_t u; float f; } v; v.u = ((uint32_t)h) << 16;
    return v.f;
}

// -------------------- Kernel 1: normalize + hi/lo bf16 split ----------------
__global__ void ntxent_normalize(const float* __restrict__ x,
                                 uint16_t* __restrict__ zhi,
                                 uint16_t* __restrict__ zlo) {
    const int wave = threadIdx.x >> 5;
    const int lane = threadIdx.x & 31;
    const int row  = blockIdx.x * 8 + wave;

    const float4 xv = ((const float4*)(x + (size_t)row * DIM))[lane];
    float ss = xv.x * xv.x + xv.y * xv.y + xv.z * xv.z + xv.w * xv.w;
#pragma unroll
    for (int off = 16; off >= 1; off >>= 1) ss += __shfl_xor(ss, off, 32);

    const float inv = 1.0f / fmaxf(sqrtf(ss), 1e-8f);
    float z[4] = { xv.x * inv, xv.y * inv, xv.z * inv, xv.w * inv };

    ushort4 hv, lv;
    uint16_t* hp = &hv.x;
    uint16_t* lp = &lv.x;
#pragma unroll
    for (int i = 0; i < 4; ++i) {
        uint16_t h = f32_to_bf16(z[i]);
        hp[i] = h;
        lp[i] = f32_to_bf16(z[i] - bf16_to_f32(h));
    }
    ((ushort4*)(zhi + (size_t)row * DIM))[lane] = hv;
    ((ushort4*)(zlo + (size_t)row * DIM))[lane] = lv;
}

// -------------------- Kernel 2: fused GEMM + fixed-shift LSE ----------------
union FragBf { v16bf v; uint16_t e[16]; uint4 q[2]; };

#define WMMA_BF16(A, B, C) \
    __builtin_amdgcn_wmma_f32_16x16x32_bf16(false, (A), false, (B), (short)0, (C), false, false)

// block = 256 threads = 8 waves. blockIdx.x = I tile (16 rows).
// Wave w sweeps J tiles [w*64, w*64+64): 64 tiles x (4 K-slabs x 3 WMMA).
__global__ void __launch_bounds__(256)
ntxent_simloss(const uint16_t* __restrict__ zhi,
               const uint16_t* __restrict__ zlo,
               float* __restrict__ row_loss) {
    __shared__ float ls_s[8][16];
    __shared__ float ls_p[8][16];

    const int wave = threadIdx.x >> 5;
    const int lane = threadIdx.x & 31;
    const int h    = lane >> 4;          // which 16-lane half
    const int l15  = lane & 15;
    const int i0   = blockIdx.x * 16;

    // ---- A fragments: rows i0..i0+15, full K=128, hi & lo (held in VGPRs).
    // A k order used here (same as B): lane m / m+16 hold row m; elems 0..15 =
    // K kb+h*16 .. +15 (32B contiguous).
    FragBf Ahi[4], Alo[4];
    {
        const size_t abase = (size_t)(i0 + l15) * DIM;
#pragma unroll
        for (int ks = 0; ks < 4; ++ks) {
            const int kb = ks * 32;
            Ahi[ks].q[0] = *(const uint4*)(zhi + abase + kb + h * 16);
            Ahi[ks].q[1] = *(const uint4*)(zhi + abase + kb + h * 16 + 8);
            Alo[ks].q[0] = *(const uint4*)(zlo + abase + kb + h * 16);
            Alo[ks].q[1] = *(const uint4*)(zlo + abase + kb + h * 16 + 8);
        }
    }

    // C slot (lane, v) holds sim[row = i0 + v + 8*h][col = j0 + l15].
    int rowid[8], poscol[8];
#pragma unroll
    for (int v = 0; v < 8; ++v) {
        rowid[v]  = i0 + v + 8 * h;
        poscol[v] = (rowid[v] + HALF_N) & (N_ROWS - 1);
    }

    float s[8], pos[8];
#pragma unroll
    for (int v = 0; v < 8; ++v) { s[v] = 0.0f; pos[v] = 0.0f; }

    // the diagonal lives only in J tile == blockIdx.x; the 16 positive-pair
    // columns for this row tile all live in J tile (blockIdx.x+256) mod 512.
    const int diag_jt = (int)blockIdx.x;
    const int pos_jt  = ((int)blockIdx.x + HALF_N / 16) & (N_TILES - 1);

    const int jt0   = wave * 64;
    const int jtEnd = jt0 + 64;

    // ---- preload slab 0 (hi+lo) of the first tile (software pipeline).
    // B k order used here (same as A): lane n / n+16 = col n; elems 0..15 =
    // K kb+h*16 .. +15 (32B contiguous).
    FragBf Bhi0, Blo0;
    {
        const size_t bb = (size_t)(jt0 * 16 + l15) * DIM;
        const uint4* bh = (const uint4*)(zhi + bb + h * 16);
        Bhi0.q[0] = bh[0]; Bhi0.q[1] = bh[1];
        const uint4* bl = (const uint4*)(zlo + bb + h * 16);
        Blo0.q[0] = bl[0]; Blo0.q[1] = bl[1];
    }

    for (int jt = jt0; jt < jtEnd; ++jt) {
        const int j0 = jt * 16;
        const size_t bbase = (size_t)(j0 + l15) * DIM;

        // slabs 1..3 for this tile
        FragBf Bhi[3], Blo[3];
#pragma unroll
        for (int ks = 1; ks < 4; ++ks) {
            const int kb = ks * 32;
            const uint4* bh = (const uint4*)(zhi + bbase + kb + h * 16);
            Bhi[ks - 1].q[0] = bh[0]; Bhi[ks - 1].q[1] = bh[1];
            const uint4* bl = (const uint4*)(zlo + bbase + kb + h * 16);
            Blo[ks - 1].q[0] = bl[0]; Blo[ks - 1].q[1] = bl[1];
        }

        v8f c = {};
        c = WMMA_BF16(Ahi[0].v, Bhi0.v, c);      // slab 0: preloaded, no stall
        c = WMMA_BF16(Ahi[0].v, Blo0.v, c);
        c = WMMA_BF16(Alo[0].v, Bhi0.v, c);
#pragma unroll
        for (int ks = 1; ks < 4; ++ks) {
            c = WMMA_BF16(Ahi[ks].v, Bhi[ks - 1].v, c);
            c = WMMA_BF16(Ahi[ks].v, Blo[ks - 1].v, c);
            c = WMMA_BF16(Alo[ks].v, Bhi[ks - 1].v, c);
        }

        // ---- prefetch slab 0 of the next tile (clamped -> loop stays
        // branch-free and in-bounds; flies during the epilogue below).
        {
            const int jtn = (jt + 1 < jtEnd) ? (jt + 1) : jt0;
            const size_t nb = (size_t)(jtn * 16 + l15) * DIM;
            const uint4* bh = (const uint4*)(zhi + nb + h * 16);
            Bhi0.q[0] = bh[0]; Bhi0.q[1] = bh[1];
            const uint4* bl = (const uint4*)(zlo + nb + h * 16);
            Blo0.q[0] = bl[0]; Blo0.q[1] = bl[1];
        }

        // ---- epilogue; special-tile guards are wave-uniform scalar branches.
        const int jts = __builtin_amdgcn_readfirstlane(jt);

        if (jts == pos_jt) {                     // capture raw positive logit
            const int col = j0 + l15;
#pragma unroll
            for (int v = 0; v < 8; ++v)
                if (col == poscol[v]) pos[v] = c[v];
        }

        float xv[8];
#pragma unroll
        for (int v = 0; v < 8; ++v) xv[v] = __builtin_fmaf(c[v], K1, -K1);

        if (jts == diag_jt) {                    // mask self-similarity
            const int col = j0 + l15;
#pragma unroll
            for (int v = 0; v < 8; ++v)
                if (col == rowid[v]) xv[v] = -__builtin_inff();  // exp2 -> 0
        }

#pragma unroll
        for (int v = 0; v < 8; ++v) s[v] += __builtin_amdgcn_exp2f(xv[v]);
    }

    // ---- plain sums across the 16 lanes sharing each row (per half).
#pragma unroll
    for (int off = 8; off >= 1; off >>= 1) {
#pragma unroll
        for (int v = 0; v < 8; ++v) {
            s[v]   += __shfl_xor(s[v], off, 32);
            pos[v] += __shfl_xor(pos[v], off, 32);
        }
    }

    if (l15 == 0) {
#pragma unroll
        for (int v = 0; v < 8; ++v) {
            ls_s[wave][v + 8 * h] = s[v];
            ls_p[wave][v + 8 * h] = pos[v];
        }
    }
    __syncthreads();

    // ---- merge the 8 wave partials per row; write per-row loss.
    //   lse = 10 + ln(S);  loss = lse - pos*10   (pos stored as raw dot)
    if (threadIdx.x < 16) {
        const int r = threadIdx.x;
        float S = 0.0f, P = 0.0f;
#pragma unroll
        for (int w = 0; w < 8; ++w) { S += ls_s[w][r]; P += ls_p[w][r]; }
        row_loss[i0 + r] = 10.0f + logf(S) - 10.0f * P;
    }
}

// -------------------- Kernel 3: deterministic mean reduction ----------------
__global__ void ntxent_reduce(const float* __restrict__ row_loss,
                              float* __restrict__ out) {
    __shared__ float acc[256];
    float t = 0.0f;
    for (int i = threadIdx.x; i < N_ROWS; i += 256) t += row_loss[i];
    acc[threadIdx.x] = t;
    __syncthreads();
#pragma unroll
    for (int off = 128; off >= 1; off >>= 1) {
        if ((int)threadIdx.x < off) acc[threadIdx.x] += acc[threadIdx.x + off];
        __syncthreads();
    }
    if (threadIdx.x == 0) out[0] = acc[0] / (float)N_ROWS;
}

// ---------------------------------------------------------------------------
extern "C" void kernel_launch(void* const* d_in, const int* in_sizes, int n_in,
                              void* d_out, int out_size, void* d_ws, size_t ws_size,
                              hipStream_t stream) {
    (void)in_sizes; (void)n_in; (void)out_size; (void)ws_size;
    const float* x = (const float*)d_in[0];
    float* out = (float*)d_out;

    // workspace: zhi (2MB) | zlo (2MB) | row_loss (32KB)
    uint16_t* zhi = (uint16_t*)d_ws;
    uint16_t* zlo = zhi + (size_t)N_ROWS * DIM;
    float* row_loss = (float*)(zlo + (size_t)N_ROWS * DIM);

    ntxent_normalize<<<N_ROWS / 8, 256, 0, stream>>>(x, zhi, zlo);
    ntxent_simloss<<<N_TILES, 256, 0, stream>>>(zhi, zlo, row_loss);
    ntxent_reduce<<<1, 256, 0, stream>>>(row_loss, out);
}
